// QuantizedLinear_47751446397344
// MI455X (gfx1250) — matrix operand order repaired and checked by reading, not compiled, not run on hardware
//
#include <hip/hip_runtime.h>

typedef _Float16     v16h __attribute__((ext_vector_type(16)));
typedef _Float16     h4   __attribute__((ext_vector_type(4)));
typedef float        v8f  __attribute__((ext_vector_type(8)));
typedef float        f4   __attribute__((ext_vector_type(4)));
typedef int          i4   __attribute__((ext_vector_type(4)));
typedef unsigned int u4   __attribute__((ext_vector_type(4)));

union FragCast { v16h v; u4 q[2]; };

#define BM 256      // block tile M
#define BN 128      // block tile N
#define BK 32       // K per stage
#define LDT 40      // LDS row stride (halfs): BK + 8 pad -> conflict-free frag reads

#if __has_builtin(__builtin_amdgcn_global_load_async_to_lds_b128)
#define HAVE_ASYNC_LDS 1
#endif

__device__ __forceinline__ void stage_b128(const _Float16* g, _Float16* l)
{
#ifdef HAVE_ASYNC_LDS
    __builtin_amdgcn_global_load_async_to_lds_b128(
        (__attribute__((address_space(1))) i4*)g,
        (__attribute__((address_space(3))) i4*)l, 0, 0);
#else
    *(u4*)l = *(const u4*)g;   // sync fallback: global_load_b128 + ds_store_b128
#endif
}

__device__ __forceinline__ void wait_async_all()
{
#ifdef HAVE_ASYNC_LDS
    asm volatile("s_wait_asynccnt 0x0" ::: "memory");
#endif
}

// ---------------- pre-pass: dtype conversion into workspace ----------------

__global__ __launch_bounds__(256)
void cvt_f32_to_f16(const float* __restrict__ in, _Float16* __restrict__ out, size_t n8)
{
    const size_t i = (size_t)blockIdx.x * 256 + threadIdx.x;
    if (i >= n8) return;
    const f4 a = *(const f4*)(in + i * 8);
    const f4 b = *(const f4*)(in + i * 8 + 4);
    *(h4*)(out + i * 8)     = __builtin_convertvector(a, h4);
    *(h4*)(out + i * 8 + 4) = __builtin_convertvector(b, h4);
}

__global__ __launch_bounds__(256)
void cvt_i32_to_f16(const int* __restrict__ in, _Float16* __restrict__ out, size_t n8)
{
    const size_t i = (size_t)blockIdx.x * 256 + threadIdx.x;
    if (i >= n8) return;
    const i4 a = *(const i4*)(in + i * 8);
    const i4 b = *(const i4*)(in + i * 8 + 4);
    *(h4*)(out + i * 8)     = __builtin_convertvector(a, h4);
    *(h4*)(out + i * 8 + 4) = __builtin_convertvector(b, h4);
}

// ---------------- fast path: f16 GEMM with async-to-LDS staging ----------------

__global__ __launch_bounds__(256)
void qlinear_wmma_async(const _Float16* __restrict__ Xh,
                        const _Float16* __restrict__ Wh,
                        const float* __restrict__ scale_p,
                        const float* __restrict__ bias,
                        float* __restrict__ Y,
                        int M, int N, int K)
{
    __shared__ _Float16 As[2][BM * LDT];   // 2 x 20 KB
    __shared__ _Float16 Bs[2][BN * LDT];   // 2 x 10 KB

    const int tid   = threadIdx.x;
    const int lane  = tid & 31;
    const int l16   = lane & 15;
    const int lhalf = lane >> 4;
    const int wave  = tid >> 5;
    const int wm    = (wave & 3) * 64;
    const int wn    = (wave >> 2) * 64;

    const int bm = blockIdx.x * BM;
    const int bn = blockIdx.y * BN;

    // staging decomposition: each thread moves 4 A-chunks + 2 B-chunks of 16B
    const int srow = tid >> 2;          // 0..63
    const int scol = (tid & 3) * 8;     // 0,8,16,24 (halfs)

    v8f acc[4][4];
    const v8f vzero = {0.f, 0.f, 0.f, 0.f, 0.f, 0.f, 0.f, 0.f};
#pragma unroll
    for (int i = 0; i < 4; ++i)
#pragma unroll
        for (int j = 0; j < 4; ++j) acc[i][j] = vzero;

    const int ntiles = K / BK;

    // prologue: issue async stage of tile 0 into buffer 0
#pragma unroll
    for (int i = 0; i < 4; ++i)
        stage_b128(Xh + (size_t)(bm + srow + i * 64) * K + scol,
                   &As[0][(srow + i * 64) * LDT + scol]);
#pragma unroll
    for (int i = 0; i < 2; ++i)
        stage_b128(Wh + (size_t)(bn + srow + i * 64) * K + scol,
                   &Bs[0][(srow + i * 64) * LDT + scol]);

    for (int t = 0; t < ntiles; ++t) {
        const int p = t & 1;

        wait_async_all();     // this wave's buffer-p writes are in LDS
        __syncthreads();      // buffer p staged by all waves; reads of p^1 all retired

        // issue async stage of tile t+1 into buffer p^1 (hidden under compute)
        if (t + 1 < ntiles) {
            const int k0 = (t + 1) * BK;
            const int q  = p ^ 1;
#pragma unroll
            for (int i = 0; i < 4; ++i)
                stage_b128(Xh + (size_t)(bm + srow + i * 64) * K + k0 + scol,
                           &As[q][(srow + i * 64) * LDT + scol]);
#pragma unroll
            for (int i = 0; i < 2; ++i)
                stage_b128(Wh + (size_t)(bn + srow + i * 64) * K + k0 + scol,
                           &Bs[q][(srow + i * 64) * LDT + scol]);
        }

        // fragments from LDS buffer p
        FragCast af[4], bf[4];
#pragma unroll
        for (int i = 0; i < 4; ++i) {
            const _Float16* pa = &As[p][(wm + i * 16 + l16) * LDT];
            af[i].q[0] = *(const u4*)(pa + lhalf * 8);
            af[i].q[1] = *(const u4*)(pa + 16 + lhalf * 8);
            const _Float16* pb = &Bs[p][(wn + i * 16 + l16) * LDT + lhalf * 8];
            bf[i].q[0] = *(const u4*)(pb);
            bf[i].q[1] = *(const u4*)(pb + 16);
        }

        // 16 WMMAs: 64x64x32 per wave per iteration
#pragma unroll
        for (int tm = 0; tm < 4; ++tm)
#pragma unroll
            for (int tn = 0; tn < 4; ++tn)
                acc[tm][tn] = __builtin_amdgcn_wmma_f32_16x16x32_f16(
                    false, af[tm].v, false, bf[tn].v,
                    (short)0, acc[tm][tn], false, false);
    }

    // epilogue: y = acc * scale + bias, non-temporal (write-once output)
    const float s = *scale_p;
#pragma unroll
    for (int tn = 0; tn < 4; ++tn) {
        const int n  = bn + wn + tn * 16 + l16;
        const float bv = bias[n];
#pragma unroll
        for (int tm = 0; tm < 4; ++tm) {
#pragma unroll
            for (int r = 0; r < 8; ++r) {
                const int m = bm + wm + tm * 16 + lhalf * 8 + r;
                __builtin_nontemporal_store(acc[tm][tn][r] * s + bv,
                                            Y + (size_t)m * N + n);
            }
        }
    }
}

// ---------------- direct path (fallback when ws too small) ----------------

__global__ __launch_bounds__(256)
void qlinear_wmma_direct(const float* __restrict__ X,
                         const int*   __restrict__ W,
                         const float* __restrict__ scale_p,
                         const float* __restrict__ bias,
                         float* __restrict__ Y,
                         int M, int N, int K)
{
    __shared__ _Float16 As[2][BM * LDT];
    __shared__ _Float16 Bs[2][BN * LDT];

    const int tid   = threadIdx.x;
    const int lane  = tid & 31;
    const int l16   = lane & 15;
    const int lhalf = lane >> 4;
    const int wave  = tid >> 5;
    const int wm    = (wave & 3) * 64;
    const int wn    = (wave >> 2) * 64;

    const int bm = blockIdx.x * BM;
    const int bn = blockIdx.y * BN;

    v8f acc[4][4];
    const v8f vzero = {0.f, 0.f, 0.f, 0.f, 0.f, 0.f, 0.f, 0.f};
#pragma unroll
    for (int i = 0; i < 4; ++i)
#pragma unroll
        for (int j = 0; j < 4; ++j) acc[i][j] = vzero;

    const int srow = tid >> 3;
    const int scol = (tid & 7) * 4;

    f4 ax[8];
    i4 bx[4];

    const int ntiles = K / BK;

#pragma unroll
    for (int i = 0; i < 8; ++i)
        ax[i] = *(const f4*)(X + (size_t)(bm + srow + i * 32) * K + scol);
#pragma unroll
    for (int i = 0; i < 4; ++i)
        bx[i] = *(const i4*)(W + (size_t)(bn + srow + i * 32) * K + scol);
#pragma unroll
    for (int i = 0; i < 8; ++i)
        *(h4*)(&As[0][(srow + i * 32) * LDT + scol]) = __builtin_convertvector(ax[i], h4);
#pragma unroll
    for (int i = 0; i < 4; ++i)
        *(h4*)(&Bs[0][(srow + i * 32) * LDT + scol]) = __builtin_convertvector(bx[i], h4);

    for (int t = 0; t < ntiles; ++t) {
        const int p = t & 1;

        if (t + 1 < ntiles) {
            const int k0 = (t + 1) * BK;
#pragma unroll
            for (int i = 0; i < 8; ++i)
                ax[i] = *(const f4*)(X + (size_t)(bm + srow + i * 32) * K + k0 + scol);
#pragma unroll
            for (int i = 0; i < 4; ++i)
                bx[i] = *(const i4*)(W + (size_t)(bn + srow + i * 32) * K + k0 + scol);
        }

        __syncthreads();

        FragCast af[4], bf[4];
#pragma unroll
        for (int i = 0; i < 4; ++i) {
            const _Float16* pa = &As[p][(wm + i * 16 + l16) * LDT];
            af[i].q[0] = *(const u4*)(pa + lhalf * 8);
            af[i].q[1] = *(const u4*)(pa + 16 + lhalf * 8);
            const _Float16* pb = &Bs[p][(wn + i * 16 + l16) * LDT + lhalf * 8];
            bf[i].q[0] = *(const u4*)(pb);
            bf[i].q[1] = *(const u4*)(pb + 16);
        }

#pragma unroll
        for (int tm = 0; tm < 4; ++tm)
#pragma unroll
            for (int tn = 0; tn < 4; ++tn)
                acc[tm][tn] = __builtin_amdgcn_wmma_f32_16x16x32_f16(
                    false, af[tm].v, false, bf[tn].v,
                    (short)0, acc[tm][tn], false, false);

        if (t + 1 < ntiles) {
            const int q = (t + 1) & 1;
#pragma unroll
            for (int i = 0; i < 8; ++i)
                *(h4*)(&As[q][(srow + i * 32) * LDT + scol]) = __builtin_convertvector(ax[i], h4);
#pragma unroll
            for (int i = 0; i < 4; ++i)
                *(h4*)(&Bs[q][(srow + i * 32) * LDT + scol]) = __builtin_convertvector(bx[i], h4);
        }
    }

    const float s = *scale_p;
#pragma unroll
    for (int tn = 0; tn < 4; ++tn) {
        const int n  = bn + wn + tn * 16 + l16;
        const float bv = bias[n];
#pragma unroll
        for (int tm = 0; tm < 4; ++tm) {
#pragma unroll
            for (int r = 0; r < 8; ++r) {
                const int m = bm + wm + tm * 16 + lhalf * 8 + r;
                __builtin_nontemporal_store(acc[tm][tn][r] * s + bv,
                                            Y + (size_t)m * N + n);
            }
        }
    }
}

extern "C" void kernel_launch(void* const* d_in, const int* in_sizes, int n_in,
                              void* d_out, int out_size, void* d_ws, size_t ws_size,
                              hipStream_t stream)
{
    const float* x     = (const float*)d_in[0];
    const int*   qw    = (const int*)d_in[1];
    const float* scale = (const float*)d_in[2];
    const float* bias  = (const float*)d_in[3];
    float*       y     = (float*)d_out;

    const int N = in_sizes[3];               // D_OUT = 4096
    const int K = in_sizes[1] / N;           // D_IN  = 4096
    const int M = in_sizes[0] / K;           // B*S   = 8192

    const size_t nx = (size_t)M * K;         // X elements
    const size_t nw = (size_t)N * K;         // W elements
    const size_t needed = (nx + nw) * sizeof(_Float16);

    dim3 grid(M / BM, N / BN);               // 32 x 32 = 1024 workgroups

    if (ws_size >= needed && (nx % 8 == 0) && (nw % 8 == 0)) {
        _Float16* Xh = (_Float16*)d_ws;
        _Float16* Wh = (_Float16*)((char*)d_ws + nx * sizeof(_Float16));

        const size_t nx8 = nx / 8, nw8 = nw / 8;
        cvt_f32_to_f16<<<(unsigned)((nx8 + 255) / 256), 256, 0, stream>>>(x, Xh, nx8);
        cvt_i32_to_f16<<<(unsigned)((nw8 + 255) / 256), 256, 0, stream>>>(qw, Wh, nw8);

        qlinear_wmma_async<<<grid, 256, 0, stream>>>(Xh, Wh, scale, bias, y, M, N, K);
    } else {
        qlinear_wmma_direct<<<grid, 256, 0, stream>>>(x, qw, scale, bias, y, M, N, K);
    }
}
